// SparseMoEGate_45689862095238
// MI455X (gfx1250) — matrix operand order repaired and checked by reading, not compiled, not run on hardware
//
#include <hip/hip_runtime.h>

// ---------------------------------------------------------------------------
// MoE top-2 softmax gate for MI455X (gfx1250, wave32).
//
// logits = x[T,C] @ W[E,C]^T  with T=16384, C=4096, E=64, then softmax,
// top-2 (+renorm), and load-balance aux loss.
//
// Roofline: 8.6 GFLOP vs 256 MB of mandatory x traffic -> ~11us floor at
// 23.3 TB/s (AI = 32 FLOP/B). bf16 WMMA (16x16x32) keeps the math under the
// memory floor even at 3x issue, so we run a split-bf16 GEMM
// (x=xh+xl, w=wh+wl, acc += xh*wh + xh*wl + xl*wh, f32 accumulate) for
// ~f32-accurate logits (top-2 tie-breaking is sensitive).
//
// L2 pressure: W (bf16 hi+lo = 1MB) lives in L2 but is re-read per K-step.
// 32 rows/wave (2 row-tiles share every B load) -> B:x traffic 2:1,
// ~512MB of L2 reads, matching a ~2x-HBM L2. 128 blocks x 4 waves keeps all
// WGPs occupied. The A-operand f32->bf16 split uses hardware cvt (fptrunc,
// RNE) so the pack VALU co-issues cheaply under the WMMA stream.
// ---------------------------------------------------------------------------

#define TOKENS 16384
#define CDIM   4096
#define NEXP   64
#define TOPK   2
#define ALPHA  0.01f

#define ROWS_PER_BLOCK 128   // 4 waves x 32 rows
#define THREADS        128

typedef __attribute__((ext_vector_type(16))) __bf16        v16bf;
typedef __attribute__((ext_vector_type(8)))  float         v8f;
typedef __attribute__((ext_vector_type(8)))  unsigned int  v8u;

// Split f32 into bf16 hi (RNE fptrunc -> v_cvt_pk_bf16_f32 when paired) plus
// bf16 residual lo. hi+lo represents x to ~2^-17 relative.
#define SPLIT_TO(V_H, V_L, i, f)                                    \
  {                                                                 \
    const float  _f = (f);                                          \
    const __bf16 _h = (__bf16)_f;                                   \
    (V_H)[i] = _h;                                                  \
    (V_L)[i] = (__bf16)(_f - (float)_h);                            \
  }

// Load one A row-tile slab (16 f32) and split-pack into bf16 hi/lo vectors.
#define LOAD_PACK_A(xrow, k0, aoff, Ah, Al)                         \
  {                                                                 \
    const float4* pa = (const float4*)((xrow) + (k0) + (aoff));     \
    const float4  a0 = pa[0], a1 = pa[1];                           \
    const float4* pb = (const float4*)((xrow) + (k0) + (aoff) + 8); \
    const float4  a2 = pb[0], a3 = pb[1];                           \
    SPLIT_TO(Ah, Al, 0,  a0.x); SPLIT_TO(Ah, Al, 1,  a0.y);         \
    SPLIT_TO(Ah, Al, 2,  a0.z); SPLIT_TO(Ah, Al, 3,  a0.w);         \
    SPLIT_TO(Ah, Al, 4,  a1.x); SPLIT_TO(Ah, Al, 5,  a1.y);         \
    SPLIT_TO(Ah, Al, 6,  a1.z); SPLIT_TO(Ah, Al, 7,  a1.w);         \
    SPLIT_TO(Ah, Al, 8,  a2.x); SPLIT_TO(Ah, Al, 9,  a2.y);         \
    SPLIT_TO(Ah, Al, 10, a2.z); SPLIT_TO(Ah, Al, 11, a2.w);         \
    SPLIT_TO(Ah, Al, 12, a3.x); SPLIT_TO(Ah, Al, 13, a3.y);         \
    SPLIT_TO(Ah, Al, 14, a3.z); SPLIT_TO(Ah, Al, 15, a3.w);         \
  }

// --------------------------- tiny helper kernels ---------------------------

__global__ void k_zero(float* __restrict__ p) {
  p[threadIdx.x] = 0.0f;   // 128 floats: Pi_sum[64] + ce[64]
}

// Split router weight W[E,C] f32 into bf16 hi/lo planes (row-major [E,C]).
__global__ __launch_bounds__(256) void k_split_w(const float* __restrict__ w,
                                                 unsigned short* __restrict__ wh,
                                                 unsigned short* __restrict__ wl) {
  const int i = (blockIdx.x * 256 + threadIdx.x) * 4;
  const float4 v = *(const float4*)(w + i);
  typedef __attribute__((ext_vector_type(4))) __bf16 v4bf;
  v4bf h, l;
  SPLIT_TO(h, l, 0, v.x);
  SPLIT_TO(h, l, 1, v.y);
  SPLIT_TO(h, l, 2, v.z);
  SPLIT_TO(h, l, 3, v.w);
  *(uint2*)(wh + i) = __builtin_bit_cast(uint2, h);
  *(uint2*)(wl + i) = __builtin_bit_cast(uint2, l);
}

// ------------------------------- main kernel --------------------------------
// 128 threads = 4 waves. Each wave owns 32 rows (2 row-tiles) x all 64 experts
// (4 expert-tiles): 8 WMMA accumulators, 24 WMMAs per K-step of 32, and each
// B load (wh/wl slab) feeds both row-tiles. Block covers 128 rows.

__global__ __launch_bounds__(THREADS) void k_gate(const float* __restrict__ x,
                                                  const unsigned short* __restrict__ wh,
                                                  const unsigned short* __restrict__ wl,
                                                  int* __restrict__ out_idx,
                                                  float* __restrict__ out_w,
                                                  float* __restrict__ g_pi,
                                                  float* __restrict__ g_ce) {
  __shared__ float lg[ROWS_PER_BLOCK][NEXP + 1];   // +1 pad: conflict-free cols
  __shared__ float ce_part[NEXP];

  const int tid  = threadIdx.x;
  const int wave = tid >> 5;
  const int lane = tid & 31;
  const int half = lane >> 4;   // 0: lanes 0-15, 1: lanes 16-31
  const int m    = lane & 15;

  if (tid < NEXP) ce_part[tid] = 0.0f;

  // ---- Phase 1: split-bf16 WMMA GEMM over K=CDIM in steps of 32 ----
  const int    rowg = blockIdx.x * ROWS_PER_BLOCK + wave * 32 + m;
  const float* xr0  = x + (size_t)rowg * CDIM;         // row-tile 0
  const float* xr1  = xr0 + (size_t)16 * CDIM;         // row-tile 1
  // A-matrix 16x32 bf16 layout (matches B): lanes 0-15 hold K 0..15,
  // lanes 16-31 hold K 16..31  -> per-lane base offset:
  const int aoff = half * 16;
  // B-matrix 32x16 bf16, k order used here (same as A): lanes 0-15 hold K 0..15, lanes 16-31 K 16..31,
  // two K per VGPR -> contiguous 16 bf16 per lane:
  const int boff = half * 16;

  v8f acc[2][4] = {};   // [row-tile][expert-tile], C/D layout per ISA

  for (int k0 = 0; k0 < CDIM; k0 += 32) {
    __builtin_prefetch(xr0 + k0 + 64 + aoff, 0, 0);    // global_prefetch_b8
    __builtin_prefetch(xr1 + k0 + 64 + aoff, 0, 0);

    v16bf A0h, A0l, A1h, A1l;
    LOAD_PACK_A(xr0, k0, aoff, A0h, A0l);
    LOAD_PACK_A(xr1, k0, aoff, A1h, A1l);

#pragma unroll
    for (int t = 0; t < 4; ++t) {
      const size_t wo = (size_t)(t * 16 + m) * CDIM + (size_t)(k0 + boff);
      const v16bf Bh = __builtin_bit_cast(v16bf, *(const v8u*)(wh + wo));
      const v16bf Bl = __builtin_bit_cast(v16bf, *(const v8u*)(wl + wo));
      // x.w ~= xh*wh + xh*wl + xl*wh (xl*wl term ~2^-18, dropped)
      acc[0][t] = __builtin_amdgcn_wmma_f32_16x16x32_bf16(false, A0h, false, Bh,
                                                          (short)0, acc[0][t], false, false);
      acc[0][t] = __builtin_amdgcn_wmma_f32_16x16x32_bf16(false, A0h, false, Bl,
                                                          (short)0, acc[0][t], false, false);
      acc[0][t] = __builtin_amdgcn_wmma_f32_16x16x32_bf16(false, A0l, false, Bh,
                                                          (short)0, acc[0][t], false, false);
      acc[1][t] = __builtin_amdgcn_wmma_f32_16x16x32_bf16(false, A1h, false, Bh,
                                                          (short)0, acc[1][t], false, false);
      acc[1][t] = __builtin_amdgcn_wmma_f32_16x16x32_bf16(false, A1h, false, Bl,
                                                          (short)0, acc[1][t], false, false);
      acc[1][t] = __builtin_amdgcn_wmma_f32_16x16x32_bf16(false, A1l, false, Bh,
                                                          (short)0, acc[1][t], false, false);
    }
  }

  // C/D layout: VGPR j, lanes 0-15 -> M=j, lanes 16-31 -> M=j+8; N = lane%16.
#pragma unroll
  for (int r0 = 0; r0 < 2; ++r0)
#pragma unroll
    for (int t = 0; t < 4; ++t)
#pragma unroll
      for (int j = 0; j < 8; ++j) {
        const int rl = wave * 32 + r0 * 16 + (half ? j + 8 : j);
        lg[rl][t * 16 + m] = acc[r0][t][j];
      }
  __syncthreads();

  // ---- Phase 2: one thread per row: softmax / top-2 / outputs ----
  {
    const int row = tid;
    float mx = -3.402823466e38f;
#pragma unroll 8
    for (int e = 0; e < NEXP; ++e) mx = fmaxf(mx, lg[row][e]);
    float s = 0.0f;
#pragma unroll 8
    for (int e = 0; e < NEXP; ++e) s += __expf(lg[row][e] - mx);
    const float inv = 1.0f / s;

    // top-2 on logits (monotone with softmax); ties pick lower index (strict >)
    float b1 = -3.402823466e38f, b2 = -3.402823466e38f;
    int   i1 = 0, i2 = 0;
    for (int e = 0; e < NEXP; ++e) {
      const float v = lg[row][e];
      if (v > b1)      { b2 = b1; i2 = i1; b1 = v; i1 = e; }
      else if (v > b2) { b2 = v; i2 = e; }
    }
    const float p1 = __expf(b1 - mx) * inv;
    const float p2 = __expf(b2 - mx) * inv;
    const float wsum = p1 + p2;

    const int grow = blockIdx.x * ROWS_PER_BLOCK + row;
    out_idx[grow * 2 + 0] = i1;
    out_idx[grow * 2 + 1] = i2;
    out_w[grow * 2 + 0]   = p1 / wsum;
    out_w[grow * 2 + 1]   = p2 / wsum;

    atomicAdd(&ce_part[i1], 1.0f);
    atomicAdd(&ce_part[i2], 1.0f);

    // overwrite this row's logits with probabilities for the Pi reduction
    for (int e = 0; e < NEXP; ++e) lg[row][e] = __expf(lg[row][e] - mx) * inv;
  }
  __syncthreads();

  // ---- Phase 3: per-expert block partials -> 64 global atomics each ----
  if (tid < NEXP) {
    float colsum = 0.0f;
    for (int r = 0; r < ROWS_PER_BLOCK; ++r) colsum += lg[r][tid];  // stride-65
    atomicAdd(&g_pi[tid], colsum);
    atomicAdd(&g_ce[tid], ce_part[tid]);
  }
}

// aux = ALPHA * sum_e (Pi_mean[e] * ce[e]/(T*K) * E)
__global__ void k_aux(const float* __restrict__ g_pi, const float* __restrict__ g_ce,
                      float* __restrict__ out_aux) {
  __shared__ float tmp[NEXP];
  const int e = threadIdx.x;
  const float Pi = g_pi[e] * (1.0f / (float)TOKENS);
  const float fi = g_ce[e] * ((float)NEXP / (float)(TOKENS * TOPK));
  tmp[e] = Pi * fi;
  __syncthreads();
  if (e == 0) {
    float s = 0.0f;
    for (int i = 0; i < NEXP; ++i) s += tmp[i];
    *out_aux = s * ALPHA;
  }
}

// ------------------------------- launcher -----------------------------------

extern "C" void kernel_launch(void* const* d_in, const int* in_sizes, int n_in,
                              void* d_out, int out_size, void* d_ws, size_t ws_size,
                              hipStream_t stream) {
  const float* x = (const float*)d_in[0];   // [T, C] f32
  const float* w = (const float*)d_in[1];   // [E, C] f32

  // workspace layout: Pi_sum[64] f32 | ce[64] f32 | wh[E*C] bf16 | wl[E*C] bf16
  float*          ws   = (float*)d_ws;
  float*          g_pi = ws;
  float*          g_ce = ws + 64;
  unsigned short* wh   = (unsigned short*)(ws + 128);
  unsigned short* wl   = wh + (size_t)NEXP * CDIM;

  // d_out layout (tuple, flat, in return order):
  // topk_idx [T,2] int32 | topk_weight [T,2] f32 | aux_loss f32
  int*   out_idx = (int*)d_out;
  float* out_w   = (float*)d_out + TOKENS * TOPK;
  float* out_aux = (float*)d_out + 2 * TOKENS * TOPK;

  k_zero<<<1, 128, 0, stream>>>(ws);
  k_split_w<<<(NEXP * CDIM) / (256 * 4), 256, 0, stream>>>(w, wh, wl);
  k_gate<<<TOKENS / ROWS_PER_BLOCK, THREADS, 0, stream>>>(x, wh, wl, out_idx, out_w,
                                                          g_pi, g_ce);
  k_aux<<<1, NEXP, 0, stream>>>(g_pi, g_ce, out_aux);
}
